// BiasFreeDenoisingGNN_57114475102686
// MI455X (gfx1250) — compile-verified, matrix operand order checked by reading, not run on hardware
//
#include <hip/hip_runtime.h>
#include <hip/hip_bf16.h>

// ---------------- problem constants ----------------
#define NN   1024
#define HH   128
#define EE   32768
#define ETOT (EE + NN)       // edges + self loops = 33792
#define LL   3
#define NTILES 64            // NN/16
#define NPAIRBLKS (NTILES * (NTILES + 1) / 2)  // 2080

typedef __bf16 bf16_t;
typedef __attribute__((ext_vector_type(16))) __bf16 v16bf;
typedef __attribute__((ext_vector_type(8)))  __bf16 v8bf;
typedef __attribute__((ext_vector_type(8)))  float  v8f;

// Load a 16x32 bf16 WMMA fragment from a row-major tile (stride in elements).
// Lane l holds row (l&15); k-chunks: [hi*8 .. hi*8+7] and [16+hi*8 .. 16+hi*8+7].
__device__ __forceinline__ v16bf load_frag_bf16(const bf16_t* tile, int stride) {
  const int lane = threadIdx.x & 31;
  const int row  = lane & 15;
  const int hi   = lane >> 4;
  const bf16_t* p = tile + row * stride + hi * 8;
  v8bf a0 = *reinterpret_cast<const v8bf*>(p);
  v8bf a1 = *reinterpret_cast<const v8bf*>(p + 16);
  v16bf r;
#pragma unroll
  for (int e = 0; e < 8; ++e) { r[e] = a0[e]; r[e + 8] = a1[e]; }
  return r;
}

// ---------------- tiny prologue kernels ----------------
__global__ void temb_kernel(const float* __restrict__ t, const float* __restrict__ tW1,
                            const float* __restrict__ tW2, float* __restrict__ temb) {
  int j = threadIdx.x;               // 128 threads
  float tv = t[0];
  float acc = 0.f;
  for (int k = 0; k < HH; ++k)
    acc += fmaxf(tv * tW1[k], 0.f) * tW2[k * HH + j];
  temb[j] = acc;
}

__global__ void init_h_kernel(const int* __restrict__ Y, const float* __restrict__ emb,
                              const float* __restrict__ temb, bf16_t* __restrict__ Xbf) {
  int idx = blockIdx.x * 256 + threadIdx.x;     // NN*HH threads
  int i = idx >> 7, f = idx & 127;
  Xbf[idx] = (bf16_t)(emb[Y[i] * HH + f] + temb[f]);
}

// BT[n*K + k] = bf16(W[k*128 + n])   (N fixed at 128)
__global__ void transpose_bf16_kernel(const float* __restrict__ W, bf16_t* __restrict__ BT, int K) {
  int idx = blockIdx.x * 256 + threadIdx.x;
  if (idx >= K * 128) return;
  int n = idx / K, k = idx - n * K;
  BT[idx] = (bf16_t)(W[k * 128 + n]);
}

// ---------------- graph kernels ----------------
__global__ void gather_kernel(const bf16_t* __restrict__ hbf, const int* __restrict__ src,
                              bf16_t* __restrict__ hs) {
  int idx = blockIdx.x * 256 + threadIdx.x;     // ETOT*16, 8 bf16 each
  int e = idx >> 4, c = idx & 15;
  if (e >= ETOT) return;
  int s = (e < EE) ? src[e] : (e - EE);
  *reinterpret_cast<v8bf*>(&hs[(size_t)e * HH + c * 8]) =
      *reinterpret_cast<const v8bf*>(&hbf[(size_t)s * HH + c * 8]);
}

__global__ void deg_kernel(const int* __restrict__ dst, float* __restrict__ deg) {
  int e = blockIdx.x * 256 + threadIdx.x;
  if (e >= ETOT) return;
  int d = (e < EE) ? dst[e] : (e - EE);
  atomicAdd(&deg[d], 1.0f);
}

__global__ void scatter_kernel(const float* __restrict__ msg, const int* __restrict__ dst,
                               float* __restrict__ agg) {
  size_t idx = (size_t)blockIdx.x * 256 + threadIdx.x;   // ETOT*HH
  int e = (int)(idx >> 7), f = (int)(idx & 127);
  if (e >= ETOT) return;
  int d = (e < EE) ? dst[e] : (e - EE);
  atomicAdd(&agg[(size_t)d * HH + f], msg[idx]);
}

__global__ void divcat_kernel(const float* __restrict__ agg, const float* __restrict__ deg,
                              const bf16_t* __restrict__ hbf, bf16_t* __restrict__ cat) {
  int idx = blockIdx.x * 256 + threadIdx.x;     // NN*HH
  int i = idx >> 7, f = idx & 127;
  cat[(size_t)i * 256 + f]       = hbf[idx];
  cat[(size_t)i * 256 + 128 + f] = (bf16_t)(agg[idx] / deg[i]);
}

// ---------------- generic WMMA GEMM:  C[M x 128] = epi(A[M x K] @ B) ----------------
// K is compile-time so all fragment arrays stay in VGPRs (no scratch spills).
// Each wave owns 2 consecutive 16-row M-tiles so every LDS B-fragment feeds 2 WMMAs.
// A bf16 row-major (lda=K); BT bf16 = B transposed, [128][K]; staged into LDS.
template <int K>
__global__ __launch_bounds__(256)
void gemm128_bf16(const bf16_t* __restrict__ A, const bf16_t* __restrict__ BT,
                  int M, float* __restrict__ Cf, bf16_t* __restrict__ Cbf,
                  const float* __restrict__ resid, int do_relu) {
  extern __shared__ bf16_t sBT[];               // 128*K bf16
  const int tid = threadIdx.x;
  constexpr int total = 128 * K;
  for (int idx = tid * 8; idx < total; idx += 256 * 8)
    *reinterpret_cast<v8bf*>(&sBT[idx]) = *reinterpret_cast<const v8bf*>(&BT[idx]);
  __syncthreads();

  const int wave = tid >> 5;
  const int lane = tid & 31;
  const int m0 = (blockIdx.x * 8 + wave) * 32;  // two 16-row tiles per wave
  if (m0 >= M) return;
  constexpr int NK = K / 32;

  v16bf a0[NK], a1[NK];
#pragma unroll
  for (int kt = 0; kt < NK; ++kt) {
    a0[kt] = load_frag_bf16(A + (size_t)m0 * K + kt * 32, K);
    a1[kt] = load_frag_bf16(A + (size_t)(m0 + 16) * K + kt * 32, K);
  }

  const int n_l = lane & 15;
  const int mb  = (lane >> 4) << 3;
#pragma unroll
  for (int nt = 0; nt < 8; ++nt) {
    v8f acc0 = {}, acc1 = {};
#pragma unroll
    for (int kt = 0; kt < NK; ++kt) {
      v16bf b = load_frag_bf16(sBT + (nt * 16) * K + kt * 32, K);
      acc0 = __builtin_amdgcn_wmma_f32_16x16x32_bf16(false, a0[kt], false, b,
                                                     (short)0, acc0, false, false);
      acc1 = __builtin_amdgcn_wmma_f32_16x16x32_bf16(false, a1[kt], false, b,
                                                     (short)0, acc1, false, false);
    }
    const int ncol = nt * 16 + n_l;
#pragma unroll
    for (int r = 0; r < 8; ++r) {
      float v = acc0[r];
      if (do_relu) v = fmaxf(v, 0.0f);
      size_t o = (size_t)(m0 + mb + r) * 128 + ncol;
      if (resid) v += resid[o];
      if (Cf)  Cf[o]  = v;
      if (Cbf) Cbf[o] = (bf16_t)v;
    }
#pragma unroll
    for (int r = 0; r < 8; ++r) {
      float v = acc1[r];
      if (do_relu) v = fmaxf(v, 0.0f);
      size_t o = (size_t)(m0 + 16 + mb + r) * 128 + ncol;
      if (resid) v += resid[o];
      if (Cf)  Cf[o]  = v;
      if (Cbf) Cbf[o] = (bf16_t)v;
    }
  }
}

// ---------------- fused pair head ----------------
// For pair (i,j), i<j:  Z = relu(U[i]+V[j]);  W = relu(Z @ eW2);  out = W @ eW3.
// One block = one 16x16 (i,j) tile; 8 waves, each wave owns 2 i-rows (16 pairs each).
__global__ __launch_bounds__(256)
void pair_head_kernel(const float* __restrict__ U, const float* __restrict__ V,
                      const bf16_t* __restrict__ e2T, const float* __restrict__ eW3,
                      float* __restrict__ out) {
  extern __shared__ char smem[];
  bf16_t* sE2 = (bf16_t*)smem;                          // 128*128 bf16   = 32768 B
  float*  sE3 = (float*)(smem + 32768);                 // 128*2 f32      =  1024 B
  bf16_t* sZ  = (bf16_t*)(smem + 32768 + 1024);         // 8*16*128 bf16  = 32768 B
  float*  sW  = (float*)(smem + 32768 + 1024 + 32768);  // 8*16*128 f32   = 65536 B

  const int tid = threadIdx.x, lane = tid & 31, wave = tid >> 5;

  for (int idx = tid * 8; idx < 128 * 128; idx += 256 * 8)
    *reinterpret_cast<v8bf*>(&sE2[idx]) = *reinterpret_cast<const v8bf*>(&e2T[idx]);
  if (tid < 256) sE3[tid] = eW3[tid];
  __syncthreads();

  // map blockIdx -> upper-triangular tile (ti, tj), tj >= ti
  int rem = blockIdx.x, ti = 0, rowlen = NTILES;
  while (rem >= rowlen) { rem -= rowlen; ++ti; --rowlen; }
  const int tj = ti + rem;
  const int i0 = ti * 16, j0 = tj * 16;

  bf16_t* zw = sZ + wave * 2048;
  float*  ww = sW + wave * 2048;

  for (int half = 0; half < 2; ++half) {
    const int li = wave * 2 + half;
    const int i  = i0 + li;

    // Z[lj][f] = relu(U[i][f] + V[j0+lj][f]) in bf16 (16 x 128)
    for (int idx = lane; idx < 16 * 128; idx += 32) {
      int lj = idx >> 7, f = idx & 127;
      float z = U[(size_t)i * HH + f] + V[(size_t)(j0 + lj) * HH + f];
      zw[idx] = (bf16_t)fmaxf(z, 0.0f);
    }
    __syncthreads();

    v16bf afrag[4];
#pragma unroll
    for (int kt = 0; kt < 4; ++kt) afrag[kt] = load_frag_bf16(zw + kt * 32, 128);

    const int n_l = lane & 15;
    const int mb  = (lane >> 4) << 3;
#pragma unroll
    for (int nt = 0; nt < 8; ++nt) {
      v8f acc = {};
#pragma unroll
      for (int kt = 0; kt < 4; ++kt) {
        v16bf b = load_frag_bf16(sE2 + (nt * 16) * 128 + kt * 32, 128);
        acc = __builtin_amdgcn_wmma_f32_16x16x32_bf16(false, afrag[kt], false, b,
                                                      (short)0, acc, false, false);
      }
      int n = nt * 16 + n_l;
#pragma unroll
      for (int r = 0; r < 8; ++r)
        ww[(mb + r) * 128 + n] = fmaxf(acc[r], 0.0f);
    }
    __syncthreads();

    // final 128->2 dot: lane -> (pair lj = lane&15, channel c = lane>>4)
    const int lj = lane & 15, c = lane >> 4;
    float dot = 0.f;
    for (int f = 0; f < 128; ++f) dot += ww[lj * 128 + f] * sE3[f * 2 + c];
    const int j = j0 + lj;
    if (i < j) {
      size_t p = (size_t)i * NN - (size_t)i * (i + 1) / 2 + (size_t)(j - i - 1);
      out[p * 2 + c] = dot;
    }
    __syncthreads();
  }
}

// ---------------- host orchestration ----------------
extern "C" void kernel_launch(void* const* d_in, const int* in_sizes, int n_in,
                              void* d_out, int out_size, void* d_ws, size_t ws_size,
                              hipStream_t stream) {
  const int*   edge_index = (const int*)  d_in[0];   // (2, E)
  const int*   Y          = (const int*)  d_in[1];
  const float* t_norm     = (const float*)d_in[2];
  const float* emb        = (const float*)d_in[3];
  const float* tW1        = (const float*)d_in[4];
  const float* tW2        = (const float*)d_in[5];
  const float* projW      = (const float*)d_in[6];
  const float* msgW1      = (const float*)d_in[7];   // (L,128,128)
  const float* msgW2      = (const float*)d_in[8];
  const float* updW       = (const float*)d_in[9];   // (L,256,128)
  const float* eW1        = (const float*)d_in[10];  // (256,128)
  const float* eW2        = (const float*)d_in[11];  // (128,128)
  const float* eW3        = (const float*)d_in[12];  // (128,2)
  const int* src = edge_index;
  const int* dst = edge_index + EE;
  float* out = (float*)d_out;

  // workspace carve-up (256B aligned)
  char* ws = (char*)d_ws;
  size_t off = 0;
  auto take = [&](size_t bytes) { char* p = ws + off; off = (off + bytes + 255) & ~(size_t)255; return p; };
  float*  temb   = (float*) take(HH * 4);
  bf16_t* Xbf    = (bf16_t*)take((size_t)NN * HH * 2);
  float*  hA_f   = (float*) take((size_t)NN * HH * 4);
  bf16_t* hA_b   = (bf16_t*)take((size_t)NN * HH * 2);
  float*  hB_f   = (float*) take((size_t)NN * HH * 4);
  bf16_t* hB_b   = (bf16_t*)take((size_t)NN * HH * 2);
  bf16_t* hs     = (bf16_t*)take((size_t)ETOT * HH * 2);
  bf16_t* tmp    = (bf16_t*)take((size_t)ETOT * HH * 2);
  float*  msgf   = (float*) take((size_t)ETOT * HH * 4);
  float*  agg    = (float*) take((size_t)NN * HH * 4);
  bf16_t* cat    = (bf16_t*)take((size_t)NN * 2 * HH * 2);
  float*  deg    = (float*) take((size_t)NN * 4);
  float*  Ubuf   = (float*) take((size_t)NN * HH * 4);
  float*  Vbuf   = (float*) take((size_t)NN * HH * 4);
  bf16_t* wT1    = (bf16_t*)take((size_t)HH * HH * 2);
  bf16_t* wT2    = (bf16_t*)take((size_t)HH * HH * 2);
  bf16_t* updT   = (bf16_t*)take((size_t)2 * HH * HH * 2);
  bf16_t* e1tT   = (bf16_t*)take((size_t)HH * HH * 2);
  bf16_t* e1bT   = (bf16_t*)take((size_t)HH * HH * 2);
  bf16_t* e2T    = (bf16_t*)take((size_t)HH * HH * 2);
  (void)ws_size; (void)in_sizes; (void)n_in; (void)out_size;

  auto gemmK128 = [&](const bf16_t* A, const bf16_t* BT, int M,
                      float* Cf, bf16_t* Cbf, const float* res, int relu) {
    gemm128_bf16<128><<<dim3((M + 255) / 256), dim3(256), 128 * 128 * 2, stream>>>(
        A, BT, M, Cf, Cbf, res, relu);
  };
  auto gemmK256 = [&](const bf16_t* A, const bf16_t* BT, int M,
                      float* Cf, bf16_t* Cbf, const float* res, int relu) {
    gemm128_bf16<256><<<dim3((M + 255) / 256), dim3(256), 128 * 256 * 2, stream>>>(
        A, BT, M, Cf, Cbf, res, relu);
  };

  // ---- prologue ----
  temb_kernel<<<1, 128, 0, stream>>>(t_norm, tW1, tW2, temb);
  init_h_kernel<<<(NN * HH) / 256, 256, 0, stream>>>(Y, emb, temb, Xbf);
  transpose_bf16_kernel<<<(HH * HH) / 256, 256, 0, stream>>>(projW, wT1, HH);
  gemmK128(Xbf, wT1, NN, hA_f, hA_b, nullptr, 1);          // h = relu(X @ projW)

  hipMemsetAsync(deg, 0, NN * 4, stream);
  deg_kernel<<<(ETOT + 255) / 256, 256, 0, stream>>>(dst, deg);

  // ---- message passing ----
  float*  hc_f = hA_f; bf16_t* hc_b = hA_b;
  float*  hn_f = hB_f; bf16_t* hn_b = hB_b;
  for (int l = 0; l < LL; ++l) {
    transpose_bf16_kernel<<<(HH * HH) / 256, 256, 0, stream>>>(msgW1 + (size_t)l * HH * HH, wT1, HH);
    transpose_bf16_kernel<<<(HH * HH) / 256, 256, 0, stream>>>(msgW2 + (size_t)l * HH * HH, wT2, HH);
    transpose_bf16_kernel<<<(2 * HH * HH) / 256, 256, 0, stream>>>(updW + (size_t)l * 2 * HH * HH, updT, 2 * HH);

    gather_kernel<<<(ETOT * 16 + 255) / 256, 256, 0, stream>>>(hc_b, src, hs);
    gemmK128(hs, wT1, ETOT, nullptr, tmp, nullptr, 1);     // relu(h_src @ msgW1)
    gemmK128(tmp, wT2, ETOT, msgf, nullptr, nullptr, 0);   // msg = .. @ msgW2
    hipMemsetAsync(agg, 0, (size_t)NN * HH * 4, stream);
    scatter_kernel<<<((size_t)ETOT * HH + 255) / 256, 256, 0, stream>>>(msgf, dst, agg);
    divcat_kernel<<<(NN * HH) / 256, 256, 0, stream>>>(agg, deg, hc_b, cat);
    gemmK256(cat, updT, NN, hn_f, hn_b, hc_f, 1);          // h = relu(cat @ updW) + h
    { float* tf = hc_f; hc_f = hn_f; hn_f = tf; }
    { bf16_t* tb = hc_b; hc_b = hn_b; hn_b = tb; }
  }

  // ---- pair head ----
  transpose_bf16_kernel<<<(HH * HH) / 256, 256, 0, stream>>>(eW1, e1tT, HH);                   // top half
  transpose_bf16_kernel<<<(HH * HH) / 256, 256, 0, stream>>>(eW1 + (size_t)HH * HH, e1bT, HH); // bottom half
  transpose_bf16_kernel<<<(HH * HH) / 256, 256, 0, stream>>>(eW2, e2T, HH);
  gemmK128(hc_b, e1tT, NN, Ubuf, nullptr, nullptr, 0);     // U = h @ eW1[:128]
  gemmK128(hc_b, e1bT, NN, Vbuf, nullptr, nullptr, 0);     // V = h @ eW1[128:]

  size_t pair_smem = 32768 + 1024 + 32768 + 65536;         // 129 KB < 320 KB WGP LDS
  pair_head_kernel<<<NPAIRBLKS, 256, pair_smem, stream>>>(Ubuf, Vbuf, e2T, eW3, out);
}
